// ConstructNet_40724879901358
// MI455X (gfx1250) — matrix operand order repaired and checked by reading, not compiled, not run on hardware
//
#include <hip/hip_runtime.h>
#include <hip/hip_bf16.h>
#include <cstddef>
#include <cstdint>

// ---------------------------------------------------------------------------
// Geometry (reference: B=4, IMSIZE=48, G0=64, G=16, D=4, C=3)
// ---------------------------------------------------------------------------
#define IMW   48
#define HW    2304              // 48*48
#define NB    4
#define P     9216              // NB*HW

typedef __attribute__((ext_vector_type(16))) __bf16 v16bf;
typedef __attribute__((ext_vector_type(8)))  __bf16 v8bf;
typedef __attribute__((ext_vector_type(8)))  float  v8f;

// ---- gfx1250 async global->LDS (16B per lane-op), tracked by ASYNCcnt ------
static __device__ inline void async_ld16(const __bf16* g, __bf16* l)
{
    asm volatile("global_load_async_to_lds_b128 %0, %1, off"
                 :: "v"((unsigned)(uintptr_t)l), "v"(g) : "memory");
}
static __device__ inline void wait_async0()
{
#if __has_builtin(__builtin_amdgcn_s_wait_asynccnt)
    __builtin_amdgcn_s_wait_asynccnt(0);
#else
    asm volatile("s_wait_asynccnt 0x0" ::: "memory");
#endif
}

// ---------------------------------------------------------------------------
// Weight packing: OIHW f32 -> bf16 in exact WMMA A-register order.
// K-order: k' = r*CPAD + ci  (r = ky*KS+kx tap index, ci padded to CPAD).
// pack[((tile*nch + chunk)*32 + lane)*16 + e], A map: K=e+8*half+(e>=8?8:0).
// ---------------------------------------------------------------------------
__global__ __launch_bounds__(256)
void pack_w_kernel(const float* __restrict__ wgt, __bf16* __restrict__ dst,
                   int cin, int ks2, int nch, int nchr, int total)
{
    int i = blockIdx.x * 256 + threadIdx.x;
    if (i >= total) return;
    int e    = i & 15;
    int L    = (i >> 4) & 31;
    int rest = i >> 9;
    int jt   = rest % nch;
    int t    = rest / nch;
    int r    = jt / nchr;
    int j    = jt - r * nchr;
    int half = L >> 4, mrow = L & 15;
    int kloc = e + (half << 3) + ((e >= 8) ? 8 : 0);     // A-layout K within chunk
    int ci   = j * 32 + kloc;
    int co   = t * 16 + mrow;
    float v  = 0.f;
    if (ci < cin) v = wgt[((size_t)co * cin + ci) * ks2 + r];
    dst[i] = (__bf16)v;
}

// ---------------------------------------------------------------------------
// Implicit-GEMM conv via V_WMMA_F32_16X16X32_BF16.
//  - activations bf16, pixel-major [P][CSIN]
//  - 4 waves/block: cotiles 4*by..4*by+3 share one 16-pixel tile
//  - per tap r: stage shifted 16 x CPAD B-tile into LDS (async), then
//    each wave runs CPAD/32 WMMA chunks from LDS.
// ---------------------------------------------------------------------------
template<int CIN, int KS, int CSIN>
__global__ __launch_bounds__(128)
void conv_wmma(const __bf16* __restrict__ in, const __bf16* __restrict__ packA,
               const float* __restrict__ bias, const __bf16* __restrict__ resid,
               int csres, __bf16* __restrict__ out, int csout, int couttiles, int relu)
{
    constexpr int CPAD  = (CIN + 31) & ~31;
    constexpr int KS2   = KS * KS;
    constexpr int NCH_R = CPAD / 32;       // chunks per tap
    constexpr int NCH   = KS2 * NCH_R;     // total chunks
    constexpr int GRP   = CPAD / 16;       // 16-channel groups per pixel

    __shared__ __attribute__((aligned(32))) __bf16 Bs[16 * CPAD];

    const int tid  = threadIdx.x;
    const int wave = tid >> 5;
    const int lane = tid & 31;
    const int col  = lane & 15;
    const int half = lane >> 4;

    const int p0     = blockIdx.x << 4;
    const int cotile = (blockIdx.y << 2) + wave;
    const bool active = cotile < couttiles;

    // tile geometry: 16 pixels on one image row (16 | 48)
    const int b  = p0 / HW;
    const int rp = p0 - b * HW;
    const int py = rp / IMW;
    const int x0 = rp - py * IMW;

    const __bf16* aptr = packA + ((size_t)cotile * NCH * 32 + lane) * 16;

    v8f acc = {};
#pragma unroll
    for (int r = 0; r < KS2; ++r) {
        const int dy = (KS == 3) ? (r / 3 - 1) : 0;
        const int dx = (KS == 3) ? (r % 3 - 1) : 0;
        const int yy = py + dy;
        if ((unsigned)yy >= (unsigned)IMW) continue;   // uniform: tap is all-zero

        // ---- cooperative stage of B tile (16 px x CPAD ch) for this tap ----
        const __bf16* rowbase = in + (size_t)(b * HW + yy * IMW) * CSIN;
        for (int idx = tid; idx < 16 * GRP; idx += 128) {
            int pix = idx / GRP;
            int g   = idx - pix * GRP;
            __bf16* dstp = &Bs[pix * CPAD + g * 16];
            int xx = x0 + dx + pix;
            if ((unsigned)xx < (unsigned)IMW) {
                const __bf16* src = rowbase + (size_t)xx * CSIN + g * 16;
                async_ld16(src,     dstp);
                async_ld16(src + 8, dstp + 8);
            } else {
                v8bf z = {};
                *(v8bf*)dstp       = z;
                *(v8bf*)(dstp + 8) = z;
            }
        }
        wait_async0();
        __syncthreads();

        // ---- consume: NCH_R WMMA chunks from LDS ----
        if (active) {
#pragma unroll
            for (int j = 0; j < NCH_R; ++j) {
                v16bf av = *(const v16bf*)(aptr + (size_t)(r * NCH_R + j) * (32 * 16));
                const __bf16* bsrc = &Bs[col * CPAD + j * 32 + (half << 3)];
                v8bf blo = *(const v8bf*)bsrc;
                v8bf bhi = *(const v8bf*)(bsrc + 16);
                v16bf bv = __builtin_shufflevector(blo, bhi,
                               0,1,2,3,4,5,6,7,8,9,10,11,12,13,14,15);
                acc = __builtin_amdgcn_wmma_f32_16x16x32_bf16(
                          false, av, false, bv, (short)0, acc, false, false);
            }
        }
        __syncthreads();
    }

    // ---- epilogue: bias / residual / relu, vectorized bf16 store ----
    if (active) {
        const int cb   = (cotile << 4) + (half << 3);   // 8 contiguous channels
        const int ppix = p0 + col;
        v8bf ov;
#pragma unroll
        for (int v = 0; v < 8; ++v) {
            float rv = acc[v];
            if (bias)  rv += bias[cb + v];
            if (resid) rv += (float)resid[(size_t)ppix * csres + cb + v];
            if (relu)  rv  = fmaxf(rv, 0.f);
            ov[v] = (__bf16)rv;
        }
        *(v8bf*)(out + (size_t)ppix * csout + cb) = ov;
    }
}

// ---------------------------------------------------------------------------
// Elementwise / reduction helpers (bf16 pixel-major activations)
// ---------------------------------------------------------------------------
__global__ __launch_bounds__(256)
void nhwc_to_xin_kernel(const float* __restrict__ ims, __bf16* __restrict__ out)
{
    int i = blockIdx.x * 256 + threadIdx.x;          // i = p*32 + c
    if (i >= P * 32) return;
    int p = i >> 5, c = i & 31;
    out[i] = (__bf16)((c < 3) ? ims[(size_t)p * 3 + c] : 0.f);
}

// training-mode BatchNorm stats over (B,H,W) per channel, input [P][64] bf16
__global__ __launch_bounds__(256)
void bn_stats_kernel(const __bf16* __restrict__ in, float* __restrict__ stats)
{
    __shared__ float ss[256], ss2[256];
    int c = blockIdx.x, t = threadIdx.x;
    float s = 0.f, s2 = 0.f;
    for (int p = t; p < P; p += 256) {
        float v = (float)in[(size_t)p * 64 + c];
        s += v; s2 += v * v;
    }
    ss[t] = s; ss2[t] = s2;
    __syncthreads();
    for (int o = 128; o > 0; o >>= 1) {
        if (t < o) { ss[t] += ss[t + o]; ss2[t] += ss2[t + o]; }
        __syncthreads();
    }
    if (t == 0) {
        float m = ss[0] * (1.f / (float)P);
        stats[c]      = m;
        stats[64 + c] = ss2[0] * (1.f / (float)P) - m * m;
    }
}

__global__ __launch_bounds__(256)
void bn_apply_relu_kernel(const __bf16* __restrict__ in, const float* __restrict__ stats,
                          const float* __restrict__ g, const float* __restrict__ bta,
                          __bf16* __restrict__ out)
{
    int i = blockIdx.x * 256 + threadIdx.x;          // i = p*64 + c
    if (i >= P * 64) return;
    int c = i & 63;
    float m = stats[c], v = stats[64 + c];
    float y = g[c] * ((float)in[i] - m) * rsqrtf(v + 1e-5f) + bta[c];
    out[i] = (__bf16)fmaxf(y, 0.f);
}

__global__ __launch_bounds__(256)
void add_relu_kernel(const __bf16* __restrict__ a, const __bf16* __restrict__ b,
                     __bf16* __restrict__ out)
{
    int i = blockIdx.x * 256 + threadIdx.x;
    if (i >= P * 64) return;
    out[i] = (__bf16)fmaxf((float)a[i] + (float)b[i], 0.f);
}

// copy 64 channels from src (stride cs) into h (stride 128), channels 0..63
__global__ __launch_bounds__(256)
void copy64_kernel(const __bf16* __restrict__ src, int cs, __bf16* __restrict__ h)
{
    int i = blockIdx.x * 256 + threadIdx.x;          // i = p*64 + c
    if (i >= P * 64) return;
    int p = i >> 6, c = i & 63;
    h[(size_t)p * 128 + c] = src[(size_t)p * cs + c];
}

__global__ __launch_bounds__(256)
void to_f32_kernel(const __bf16* __restrict__ in, float* __restrict__ out)
{
    int i = blockIdx.x * 256 + threadIdx.x;
    if (i >= P * 64) return;
    out[i] = (float)in[i];
}

// w[b,e] = sigmoid( dot64( ks[b,src[e]], qs[b,dst[e]] ) ); ks/qs bf16 [P][64]
__global__ __launch_bounds__(256)
void edge_dot_kernel(const __bf16* __restrict__ ks, const __bf16* __restrict__ qs,
                     const int* __restrict__ edges, float* __restrict__ out, int E)
{
    int i = blockIdx.x * 256 + threadIdx.x;
    if (i >= NB * E) return;
    int b = i / E, e = i - b * E;
    const __bf16* kp = ks + ((size_t)b * HW + edges[e])     * 64;
    const __bf16* qp = qs + ((size_t)b * HW + edges[E + e]) * 64;
    float acc = 0.f;
#pragma unroll
    for (int g = 0; g < 8; ++g) {
        v8bf ka = *(const v8bf*)(kp + g * 8);
        v8bf qa = *(const v8bf*)(qp + g * 8);
#pragma unroll
        for (int d = 0; d < 8; ++d) acc += (float)ka[d] * (float)qa[d];
    }
    out[i] = 1.f / (1.f + __expf(-acc));
}

// ---------------------------------------------------------------------------
// Host-side orchestration
// ---------------------------------------------------------------------------
extern "C" void kernel_launch(void* const* d_in, const int* in_sizes, int n_in,
                              void* d_out, int out_size, void* d_ws, size_t ws_size,
                              hipStream_t stream)
{
    (void)n_in; (void)out_size; (void)ws_size;
    // input unpack: ims, params leaves (JAX sorted-key order), edges
    const float* ims      = (const float*)d_in[0];
    const float* gff1_b   = (const float*)d_in[1];
    const float* gff1_w   = (const float*)d_in[2];
    const float* gff2_b   = (const float*)d_in[3];
    const float* gff2_w   = (const float*)d_in[4];
    const float* k_proj_b = (const float*)d_in[5];
    const float* k_proj_w = (const float*)d_in[6];
    const float* k_bn1_b  = (const float*)d_in[7];
    const float* k_bn1_g  = (const float*)d_in[8];
    const float* k_bn2_b  = (const float*)d_in[9];
    const float* k_bn2_g  = (const float*)d_in[10];
    const float* k_c1_w   = (const float*)d_in[11];
    const float* k_c2_w   = (const float*)d_in[12];
    const float* q_proj_b = (const float*)d_in[13];
    const float* q_proj_w = (const float*)d_in[14];
    const float* q_bn1_b  = (const float*)d_in[15];
    const float* q_bn1_g  = (const float*)d_in[16];
    const float* q_bn2_b  = (const float*)d_in[17];
    const float* q_bn2_g  = (const float*)d_in[18];
    const float* q_c1_w   = (const float*)d_in[19];
    const float* q_c2_w   = (const float*)d_in[20];
    const float* sfe1_b   = (const float*)d_in[53];
    const float* sfe1_w   = (const float*)d_in[54];
    const float* sfe2_b   = (const float*)d_in[55];
    const float* sfe2_w   = (const float*)d_in[56];
    const int*   edges    = (const int*)d_in[57];
    const int    E        = in_sizes[57] / 2;

    // workspace bump allocator (bf16 units; all sizes multiples of 16 -> 32B align)
    __bf16* ws = (__bf16*)d_ws;
    size_t off = 0;
    auto alloc = [&](size_t n) { __bf16* r = ws + off; off += (n + 15) & ~(size_t)15; return r; };

    __bf16* xin   = alloc((size_t)32  * P);
    __bf16* f1    = alloc((size_t)64  * P);
    __bf16* xbuf  = alloc((size_t)64  * P);
    __bf16* h     = alloc((size_t)128 * P);
    __bf16* outs  = alloc((size_t)256 * P);
    __bf16* tmp1  = alloc((size_t)64  * P);
    __bf16* feats = alloc((size_t)64  * P);
    __bf16* y1    = alloc((size_t)64  * P);
    __bf16* y2    = alloc((size_t)64  * P);
    __bf16* kcm   = alloc((size_t)64  * P);
    __bf16* qcm   = alloc((size_t)64  * P);

    auto packsize = [](int cout, int cin, int ks) {
        int cpad = (cin + 31) & ~31;
        return (size_t)(cout / 16) * (ks * ks * (cpad / 32)) * 512;
    };
    auto pack = [&](const float* w, int cout, int cin, int ks) -> __bf16* {
        int cpad = (cin + 31) & ~31;
        int nchr = cpad / 32;
        int nch  = ks * ks * nchr;
        int total = (cout / 16) * nch * 512;
        __bf16* dst = alloc((size_t)total);
        pack_w_kernel<<<(total + 255) / 256, 256, 0, stream>>>(w, dst, cin, ks * ks, nch, nchr, total);
        return dst;
    };

    // pack all layer weights (deterministic, every call)
    __bf16* pk_sfe1 = pack(sfe1_w, 64, 3, 3);
    __bf16* pk_sfe2 = pack(sfe2_w, 64, 64, 3);
    __bf16 *pk_cw0[4], *pk_cw1[4], *pk_cw2[4], *pk_lff[4];
    for (int d = 0; d < 4; ++d) {
        pk_cw0[d] = pack((const float*)d_in[21 + d * 8 + 3], 16, 64, 3);
        pk_cw1[d] = pack((const float*)d_in[21 + d * 8 + 4], 16, 80, 3);
        pk_cw2[d] = pack((const float*)d_in[21 + d * 8 + 5], 16, 96, 3);
        pk_lff[d] = pack((const float*)d_in[21 + d * 8 + 7], 64, 112, 1);
    }
    __bf16* pk_gff1 = pack(gff1_w, 64, 256, 1);
    __bf16* pk_gff2 = pack(gff2_w, 64, 64, 3);
    __bf16* pk_kc1  = pack(k_c1_w, 64, 64, 3);
    __bf16* pk_kc2  = pack(k_c2_w, 64, 64, 3);
    __bf16* pk_kpj  = pack(k_proj_w, 64, 64, 1);
    __bf16* pk_qc1  = pack(q_c1_w, 64, 64, 3);
    __bf16* pk_qc2  = pack(q_c2_w, 64, 64, 3);
    __bf16* pk_qpj  = pack(q_proj_w, 64, 64, 1);
    (void)packsize;

    // f32 stats region (32B-aligned)
    float* stats = (float*)(ws + off);

    auto cgrid = [](int couttiles) { return dim3(P / 16, (couttiles + 3) / 4); };
    const int EB = (P * 64 + 255) / 256;

    // --- RDN backbone ---
    nhwc_to_xin_kernel<<<(P * 32 + 255) / 256, 256, 0, stream>>>(ims, xin);
    conv_wmma<3, 3, 32><<<cgrid(4), 128, 0, stream>>>(xin, pk_sfe1, sfe1_b, nullptr, 0, f1, 64, 4, 0);
    conv_wmma<64, 3, 64><<<cgrid(4), 128, 0, stream>>>(f1, pk_sfe2, sfe2_b, nullptr, 0, xbuf, 64, 4, 0);

    const __bf16* xcur = xbuf;
    int cs_x = 64;
    for (int d = 0; d < 4; ++d) {
        const float* cb0  = (const float*)d_in[21 + d * 8 + 0];
        const float* cb1  = (const float*)d_in[21 + d * 8 + 1];
        const float* cb2  = (const float*)d_in[21 + d * 8 + 2];
        const float* lffb = (const float*)d_in[21 + d * 8 + 6];

        copy64_kernel<<<EB, 256, 0, stream>>>(xcur, cs_x, h);
        conv_wmma<64, 3, 128><<<cgrid(1), 128, 0, stream>>>(h, pk_cw0[d], cb0, nullptr, 0, h + 64, 128, 1, 1);
        conv_wmma<80, 3, 128><<<cgrid(1), 128, 0, stream>>>(h, pk_cw1[d], cb1, nullptr, 0, h + 80, 128, 1, 1);
        conv_wmma<96, 3, 128><<<cgrid(1), 128, 0, stream>>>(h, pk_cw2[d], cb2, nullptr, 0, h + 96, 128, 1, 1);
        __bf16* xo = outs + d * 64;                         // channels [64d,64d+64) of [P][256]
        conv_wmma<112, 1, 128><<<cgrid(4), 128, 0, stream>>>(h, pk_lff[d], lffb, xcur, cs_x, xo, 256, 4, 0);
        xcur = xo; cs_x = 256;
    }
    conv_wmma<256, 1, 256><<<cgrid(4), 128, 0, stream>>>(outs, pk_gff1, gff1_b, nullptr, 0, tmp1, 64, 4, 0);
    conv_wmma<64, 3, 64><<<cgrid(4), 128, 0, stream>>>(tmp1, pk_gff2, gff2_b, f1, 64, feats, 64, 4, 0);

    // --- k/q heads ---
    auto kq = [&](__bf16* c1, __bf16* c2, const float* g1, const float* b1,
                  const float* g2, const float* b2, __bf16* pj, const float* pb, __bf16* cm) {
        conv_wmma<64, 3, 64><<<cgrid(4), 128, 0, stream>>>(feats, c1, nullptr, nullptr, 0, y1, 64, 4, 0);
        bn_stats_kernel<<<64, 256, 0, stream>>>(y1, stats);
        bn_apply_relu_kernel<<<EB, 256, 0, stream>>>(y1, stats, g1, b1, y1);
        conv_wmma<64, 3, 64><<<cgrid(4), 128, 0, stream>>>(y1, c2, nullptr, nullptr, 0, y2, 64, 4, 0);
        bn_stats_kernel<<<64, 256, 0, stream>>>(y2, stats);
        bn_apply_relu_kernel<<<EB, 256, 0, stream>>>(y2, stats, g2, b2, y2);
        add_relu_kernel<<<EB, 256, 0, stream>>>(feats, y2, y1);
        conv_wmma<64, 1, 64><<<cgrid(4), 128, 0, stream>>>(y1, pj, pb, nullptr, 0, cm, 64, 4, 0);
    };
    kq(pk_kc1, pk_kc2, k_bn1_g, k_bn1_b, k_bn2_g, k_bn2_b, pk_kpj, k_proj_b, kcm);
    kq(pk_qc1, pk_qc2, q_bn1_g, q_bn1_b, q_bn2_g, q_bn2_b, pk_qpj, q_proj_b, qcm);

    // --- outputs: w[4,E] then node_feats[4,2304,64] ---
    float* w_out  = (float*)d_out;
    float* nf_out = w_out + (size_t)NB * E;
    edge_dot_kernel<<<((size_t)NB * E + 255) / 256, 256, 0, stream>>>(kcm, qcm, edges, w_out, E);
    to_f32_kernel<<<EB, 256, 0, stream>>>(feats, nf_out);
}
